// GraphNN_14190571946081
// MI455X (gfx1250) — matrix operand order repaired and checked by reading, not compiled, not run on hardware
//
#include <hip/hip_runtime.h>
#include <hip/hip_bf16.h>
#include <math.h>

#define NNODES 20000
#define NEDGES 320000
#define HID    256
#define NPREV  (NNODES / 2)
#define EPSV   1e-5f
#define BOOSTF 1.5f

typedef __attribute__((ext_vector_type(16))) _Float16 v16h;
typedef __attribute__((ext_vector_type(8)))  float    v8f;
typedef int v4i_t __attribute__((vector_size(16)));

union FragU {
    v16h  v;
    uint4 q[2];
};

// ---- gfx1250 async global->LDS copy path (hedged: falls back if builtins absent) ----
#if __has_builtin(__builtin_amdgcn_global_load_async_to_lds_b128) && \
    __has_builtin(__builtin_amdgcn_s_wait_asynccnt)
#define USE_ASYNC 1
#define ASYNC_COPY16(gp, lp)                                                   \
    __builtin_amdgcn_global_load_async_to_lds_b128(                            \
        (__attribute__((address_space(1))) v4i_t*)(gp),                        \
        (__attribute__((address_space(3))) v4i_t*)(lp), 0, 0)
#define ASYNC_WAIT() __builtin_amdgcn_s_wait_asynccnt(0)
#else
#define USE_ASYNC 0
#endif

__device__ __forceinline__ void atomAddF(float* p, float v) {
    unsafeAtomicAdd(p, v);
}

// ---------------------------------------------------------------------------
// Weight prep: WT[n][k] = W[k][n] * bnscale(n)  (f16), FB[n] = folded bias
// ---------------------------------------------------------------------------
__global__ void prep_weight(const float* __restrict__ W, const float* __restrict__ b,
                            const float* __restrict__ g, const float* __restrict__ bb,
                            const float* __restrict__ m, const float* __restrict__ v,
                            _Float16* __restrict__ WT, float* __restrict__ FB,
                            int K, int NC)
{
    int idx = blockIdx.x * blockDim.x + threadIdx.x;
    int total = K * NC;
    if (idx < total) {
        int k = idx / NC, n = idx % NC;
        float s = g ? g[n] * rsqrtf(v[n] + EPSV) : 1.0f;
        WT[(size_t)n * K + k] = (_Float16)(W[idx] * s);
    }
    if (idx < NC) {
        float s = g ? g[idx] * rsqrtf(v[idx] + EPSV) : 1.0f;
        FB[idx] = g ? (b[idx] - m[idx]) * s + bb[idx] : b[idx];
    }
}

__global__ void f32_to_f16_k(const float* __restrict__ in, _Float16* __restrict__ out, int n)
{
    int i = blockIdx.x * blockDim.x + threadIdx.x;
    if (i < n) out[i] = (_Float16)in[i];
}

__global__ void fill_f32_k(float* __restrict__ p, float val, int n)
{
    int i = blockIdx.x * blockDim.x + threadIdx.x;
    if (i < n) p[i] = val;
}

// ---------------------------------------------------------------------------
// Generic WMMA GEMM: out = act(A[M,K] @ WT^T + bias)
//   A  : f16 row-major [M][K]
//   WT : f16 [NC][K]  (transposed weights, BN-folded)
//   block = 128 threads (4 waves); block tile 64(M) x 64(N); grid.y = NC/64
//   A tile double-buffered in LDS; async copy engine fills next tile while
//   WMMAs consume the current one.
// ---------------------------------------------------------------------------
template<int K, int NC, bool RELU>
__global__ void __launch_bounds__(128)
gemm_wmma(const _Float16* __restrict__ A, const _Float16* __restrict__ WT,
          const float* __restrict__ bias,
          float* __restrict__ outF, _Float16* __restrict__ outH, int M)
{
    __shared__ _Float16 As[2][64][32];
    const int tid  = threadIdx.x;
    const int wave = tid >> 5;
    const int lane = tid & 31;
    const int half = lane >> 4;
    const int nIn  = lane & 15;
    const int rowBlock = blockIdx.x * 64;
    const int colBase  = blockIdx.y * 64;
    constexpr int STEPS = K / 32;

    // per-thread staging coords: two 16B chunks per thread per 64x32 tile
    const int r0 = tid >> 2;
    const int c0 = (tid & 3) * 8;
    const int r1 = (tid + 128) >> 2;
    const int c1 = ((tid + 128) & 3) * 8;

    auto stage = [&](int s, int buf) {
        const int k0 = s * 32;
        const int gr0 = rowBlock + r0, gr1 = rowBlock + r1;
#if USE_ASYNC
        // rows >= M are not staged; their C rows are masked in the epilogue
        if (gr0 < M) ASYNC_COPY16(A + (size_t)gr0 * K + k0 + c0, &As[buf][r0][c0]);
        if (gr1 < M) ASYNC_COPY16(A + (size_t)gr1 * K + k0 + c1, &As[buf][r1][c1]);
#else
        uint4 v0 = make_uint4(0u, 0u, 0u, 0u), v1 = make_uint4(0u, 0u, 0u, 0u);
        if (gr0 < M) v0 = *(const uint4*)(A + (size_t)gr0 * K + k0 + c0);
        if (gr1 < M) v1 = *(const uint4*)(A + (size_t)gr1 * K + k0 + c1);
        *(uint4*)(&As[buf][r0][c0]) = v0;
        *(uint4*)(&As[buf][r1][c1]) = v1;
#endif
    };

    v8f acc[4] = {};
    stage(0, 0);

    for (int s = 0; s < STEPS; ++s) {
        const int buf = s & 1;
        const int k0  = s * 32;
#if USE_ASYNC
        ASYNC_WAIT();           // my copies for tile s are done
#endif
        __syncthreads();        // everyone's copies visible
        if (s + 1 < STEPS) stage(s + 1, buf ^ 1);

        // A fragment (ISA 16-bit 16x32 layout): lanes0-15 K[0..7]+[16..23],
        // lanes16-31 K[8..15]+[24..31]
        FragU af;
        {
            int r = wave * 16 + nIn;
            af.q[0] = *(const uint4*)(&As[buf][r][half * 8]);
            af.q[1] = *(const uint4*)(&As[buf][r][half * 8 + 16]);
        }
        #pragma unroll
        for (int c = 0; c < 4; ++c) {
            FragU bf;
            const _Float16* bp = WT + (size_t)(colBase + c * 16 + nIn) * K + k0 + half * 8;
            bf.q[0] = *(const uint4*)(bp);
            bf.q[1] = *(const uint4*)(bp + 16);
            __builtin_prefetch(bp + 32, 0, 0);   // next k-step's B fragment
            acc[c] = __builtin_amdgcn_wmma_f32_16x16x32_f16(
                false, af.v, false, bf.v, (short)0, acc[c], false, false);
        }
        __syncthreads();        // protect As[buf] before it is re-staged at s+2
    }

    // epilogue: C layout lane<16 -> M=r, lane>=16 -> M=r+8, N = lane&15
    #pragma unroll
    for (int c = 0; c < 4; ++c) {
        int gn  = colBase + c * 16 + nIn;
        float b = bias ? bias[gn] : 0.0f;
        #pragma unroll
        for (int r = 0; r < 8; ++r) {
            int gm = rowBlock + wave * 16 + r + half * 8;
            if (gm < M) {
                float vv = acc[c][r] + b;
                if (RELU) vv = fmaxf(vv, 0.0f);
                if (outF) outF[(size_t)gm * NC + gn] = vv;
                if (outH) outH[(size_t)gm * NC + gn] = (_Float16)vv;
            }
        }
    }
}

// ---------------------------------------------------------------------------
// Fused edge head: sigmoid(L3(relu(L2(relu(BN(L1(concat(x[s],x[d]))))))))
//   16 edges per block, 256 threads (8 waves). Whole MLP in LDS via WMMA.
// ---------------------------------------------------------------------------
__global__ void __launch_bounds__(256)
edge_head(const _Float16* __restrict__ X,                       // [N][256] f16
          const int* __restrict__ srcIdx, const int* __restrict__ dstIdx,
          const _Float16* __restrict__ W1T, const float* __restrict__ fb1,  // [256][512]
          const _Float16* __restrict__ W2T, const float* __restrict__ fb2,  // [128][256]
          const float* __restrict__ w3, const float* __restrict__ b3,       // [128],[1]
          float* __restrict__ out, int nE)
{
    __shared__ _Float16 EF[16][512];
    __shared__ _Float16 H1[16][256];
    __shared__ _Float16 H2[16][128];
    __shared__ int SS[16], DD[16];

    const int tid  = threadIdx.x;
    const int wave = tid >> 5;
    const int lane = tid & 31;
    const int half = lane >> 4;
    const int nIn  = lane & 15;
    const int e0   = blockIdx.x * 16;

    if (tid < 16) {
        int e = e0 + tid;
        SS[tid] = (e < nE) ? srcIdx[e] : 0;
        DD[tid] = (e < nE) ? dstIdx[e] : 0;
    }
    __syncthreads();

    // gather concat(x[src],x[dst]) -> EF : 1024 16B chunks, 4 per thread
    #pragma unroll
    for (int i = 0; i < 4; ++i) {
        int q    = tid + i * 256;
        int e    = q >> 6;
        int rsel = (q >> 5) & 1;
        int off  = (q & 31) * 8;
        int row  = rsel ? DD[e] : SS[e];
        const _Float16* gp = X + (size_t)row * 256 + off;
        _Float16* lp = &EF[e][rsel * 256 + off];
#if USE_ASYNC
        ASYNC_COPY16(gp, lp);
#else
        *(uint4*)lp = *(const uint4*)gp;
#endif
    }
#if USE_ASYNC
    ASYNC_WAIT();
#endif
    __syncthreads();

    // stage 1: H1 = relu(EF @ W1T^T + fb1)  [16 x 256]; wave owns cols [32w,32w+32)
    {
        v8f acc[2] = {};
        for (int k0 = 0; k0 < 512; k0 += 32) {
            FragU af;
            af.q[0] = *(const uint4*)(&EF[nIn][k0 + half * 8]);
            af.q[1] = *(const uint4*)(&EF[nIn][k0 + half * 8 + 16]);
            #pragma unroll
            for (int c = 0; c < 2; ++c) {
                int col = wave * 32 + c * 16 + nIn;
                FragU bf;
                const _Float16* bp = W1T + (size_t)col * 512 + k0 + half * 8;
                bf.q[0] = *(const uint4*)bp;
                bf.q[1] = *(const uint4*)(bp + 16);
                __builtin_prefetch(bp + 32, 0, 0);
                acc[c] = __builtin_amdgcn_wmma_f32_16x16x32_f16(
                    false, af.v, false, bf.v, (short)0, acc[c], false, false);
            }
        }
        #pragma unroll
        for (int c = 0; c < 2; ++c) {
            int col = wave * 32 + c * 16 + nIn;
            float b = fb1[col];
            #pragma unroll
            for (int r = 0; r < 8; ++r)
                H1[r + half * 8][col] = (_Float16)fmaxf(acc[c][r] + b, 0.0f);
        }
    }
    __syncthreads();

    // stage 2: H2 = relu(H1 @ W2T^T + fb2)  [16 x 128]; wave owns col tile w
    {
        v8f acc = {};
        for (int k0 = 0; k0 < 256; k0 += 32) {
            FragU af;
            af.q[0] = *(const uint4*)(&H1[nIn][k0 + half * 8]);
            af.q[1] = *(const uint4*)(&H1[nIn][k0 + half * 8 + 16]);
            int col = wave * 16 + nIn;
            FragU bf;
            const _Float16* bp = W2T + (size_t)col * 256 + k0 + half * 8;
            bf.q[0] = *(const uint4*)bp;
            bf.q[1] = *(const uint4*)(bp + 16);
            acc = __builtin_amdgcn_wmma_f32_16x16x32_f16(
                false, af.v, false, bf.v, (short)0, acc, false, false);
        }
        int col = wave * 16 + nIn;
        float b = fb2[col];
        #pragma unroll
        for (int r = 0; r < 8; ++r)
            H2[r + half * 8][col] = (_Float16)fmaxf(acc[r] + b, 0.0f);
    }
    __syncthreads();

    // stage 3: per-edge dot(H2[e], w3) + b3 -> sigmoid. wave handles 2 edges.
    #pragma unroll
    for (int i = 0; i < 2; ++i) {
        int e = wave * 2 + i;
        float p = 0.0f;
        for (int t = lane; t < 128; t += 32) p += (float)H2[e][t] * w3[t];
        #pragma unroll
        for (int m = 16; m >= 1; m >>= 1) p += __shfl_xor(p, m, 32);
        if (lane == 0 && (e0 + e) < nE)
            out[e0 + e] = 1.0f / (1.0f + expf(-(p + b3[0])));
    }
}

// ---------------------------------------------------------------------------
// GCN helpers
// ---------------------------------------------------------------------------
__global__ void deg_acc_k(const int* __restrict__ dst, float* __restrict__ deg, int nE)
{
    int e = blockIdx.x * blockDim.x + threadIdx.x;
    if (e < nE) atomAddF(&deg[dst[e]], 1.0f);
}

__global__ void rsqrt_k(const float* __restrict__ deg, float* __restrict__ dinv, int n)
{
    int i = blockIdx.x * blockDim.x + threadIdx.x;
    if (i < n) dinv[i] = rsqrtf(deg[i]);
}

__global__ void gcn_scatter(const float* __restrict__ xw,
                            const int* __restrict__ src, const int* __restrict__ dst,
                            const float* __restrict__ dinv, float* __restrict__ agg, int nE)
{
    int idx = blockIdx.x * blockDim.x + threadIdx.x;
    if (idx >= nE * HID) return;
    int e = idx >> 8;
    int c = idx & 255;
    int s = src[e], d = dst[e];
    float w = dinv[s] * dinv[d];
    atomAddF(&agg[(size_t)d * HID + c], w * xw[(size_t)s * HID + c]);
}

__global__ void gcn_finalize(const float* __restrict__ agg, const float* __restrict__ xw,
                             const float* __restrict__ prev, const float* __restrict__ dinv,
                             const float* __restrict__ bias,
                             float* __restrict__ outF, _Float16* __restrict__ outH, int n)
{
    int idx = blockIdx.x * blockDim.x + threadIdx.x;
    if (idx >= n * HID) return;
    int i = idx >> 8, c = idx & 255;
    float di = dinv[i];
    float v = agg[idx] + di * di * xw[idx] + bias[c] + prev[idx];
    if (outF) outF[idx] = v;
    if (outH) outH[idx] = (_Float16)v;
}

__global__ void boost_k(const float* __restrict__ ewb, const float* __restrict__ attn,
                        const int* __restrict__ src, const int* __restrict__ dst,
                        float* __restrict__ out, int nE)
{
    int e = blockIdx.x * blockDim.x + threadIdx.x;
    if (e >= nE) return;
    bool sm = src[e] < NPREV, dm = dst[e] < NPREV;
    float w = ewb[e];
    if (sm != dm) { float a = attn[e]; w = w * (1.0f + BOOSTF * a * a); }
    out[e] = w;
}

// ---------------------------------------------------------------------------
static inline int ceilDiv(int a, int b) { return (a + b - 1) / b; }

extern "C" void kernel_launch(void* const* d_in, const int* in_sizes, int n_in,
                              void* d_out, int out_size, void* d_ws, size_t ws_size,
                              hipStream_t stream)
{
    (void)in_sizes; (void)n_in; (void)out_size; (void)ws_size;

    // --- input pointers (setup_inputs dict order, depth-first) ---
    const float* nf      = (const float*)d_in[0];
    const float* W_proj  = (const float*)d_in[1];  const float* b_proj = (const float*)d_in[2];
    const float* W_enc1  = (const float*)d_in[3];  const float* b_enc1 = (const float*)d_in[4];
    const float* g1 = (const float*)d_in[5], *bb1 = (const float*)d_in[6],
               * m1 = (const float*)d_in[7], *v1  = (const float*)d_in[8];
    const float* W_enc2  = (const float*)d_in[9];  const float* b_enc2 = (const float*)d_in[10];
    const float* g2 = (const float*)d_in[11], *bb2 = (const float*)d_in[12],
               * m2 = (const float*)d_in[13], *v2  = (const float*)d_in[14];
    const float* W_g0 = (const float*)d_in[15]; const float* b_g0 = (const float*)d_in[16];
    const float* W_g1 = (const float*)d_in[17]; const float* b_g1 = (const float*)d_in[18];
    const float* W_ta1 = (const float*)d_in[19]; const float* b_ta1 = (const float*)d_in[20];
    const float* gt = (const float*)d_in[21], *bbt = (const float*)d_in[22],
               * mt = (const float*)d_in[23], *vt  = (const float*)d_in[24];
    const float* W_ta2 = (const float*)d_in[25]; const float* b_ta2 = (const float*)d_in[26];
    const float* W_ta3 = (const float*)d_in[27]; const float* b_ta3 = (const float*)d_in[28];
    const float* W_ep1 = (const float*)d_in[29]; const float* b_ep1 = (const float*)d_in[30];
    const float* ge = (const float*)d_in[31], *bbe = (const float*)d_in[32],
               * me = (const float*)d_in[33], *ve  = (const float*)d_in[34];
    const float* W_ep2 = (const float*)d_in[35]; const float* b_ep2 = (const float*)d_in[36];
    const float* W_ep3 = (const float*)d_in[37]; const float* b_ep3 = (const float*)d_in[38];
    const float* W_id1 = (const float*)d_in[39]; const float* b_id1 = (const float*)d_in[40];
    const float* W_id2 = (const float*)d_in[41]; const float* b_id2 = (const float*)d_in[42];
    const int*   ei    = (const int*)d_in[43];
    const int*   src   = ei;
    const int*   dst   = ei + NEDGES;

    // --- workspace layout ---
    char* ws = (char*)d_ws;
    size_t off = 0;
    auto A = [&](size_t bytes) -> char* {
        char* p = ws + off;
        off = (off + bytes + 255) & ~(size_t)255;
        return p;
    };
    const size_t NB = (size_t)NNODES * HID * sizeof(float);
    float* B0f = (float*)A(NB);  _Float16* B0h = (_Float16*)B0f;
    float* B1f = (float*)A(NB);  _Float16* B1h = (_Float16*)B1f;
    float* B2f = (float*)A(NB);  _Float16* B2h = (_Float16*)B2f;
    float* B3f = (float*)A(NB);
    float* B4f = (float*)A(NB);
    float* attnB = (float*)A((size_t)NEDGES * 4);
    float* ewB   = (float*)A((size_t)NEDGES * 4);
    float* degB  = (float*)A((size_t)NNODES * 4);
    float* dinvB = (float*)A((size_t)NNODES * 4);

    _Float16* projWT = (_Float16*)A(256 * 128 * 2); float* projFB = (float*)A(128 * 4);
    _Float16* enc1WT = (_Float16*)A(128 * 256 * 2); float* enc1FB = (float*)A(256 * 4);
    _Float16* enc2WT = (_Float16*)A(256 * 256 * 2); float* enc2FB = (float*)A(256 * 4);
    _Float16* g0WT   = (_Float16*)A(256 * 256 * 2); float* g0FB   = (float*)A(256 * 4);
    _Float16* g1WT   = (_Float16*)A(256 * 256 * 2); float* g1FB   = (float*)A(256 * 4);
    _Float16* ta1WT  = (_Float16*)A(512 * 256 * 2); float* ta1FB  = (float*)A(256 * 4);
    _Float16* ta2WT  = (_Float16*)A(256 * 128 * 2); float* ta2FB  = (float*)A(128 * 4);
    _Float16* ep1WT  = (_Float16*)A(512 * 256 * 2); float* ep1FB  = (float*)A(256 * 4);
    _Float16* ep2WT  = (_Float16*)A(256 * 128 * 2); float* ep2FB  = (float*)A(128 * 4);
    _Float16* id1WT  = (_Float16*)A(256 * 256 * 2); float* id1FB  = (float*)A(256 * 4);
    _Float16* id2WT  = (_Float16*)A(256 * 128 * 2); float* id2FB  = (float*)A(128 * 4);

    float* outX  = (float*)d_out;                              // [N,256]
    float* outEW = outX + (size_t)NNODES * HID;                // [E]
    float* outID = outEW + NEDGES;                             // [N,128]

    // --- weight prep (BN-fold + transpose to f16) ---
    auto prep = [&](const float* W, const float* b, const float* g, const float* bb,
                    const float* m, const float* v, _Float16* WT, float* FB, int K, int NC) {
        prep_weight<<<ceilDiv(K * NC, 256), 256, 0, stream>>>(W, b, g, bb, m, v, WT, FB, K, NC);
    };
    prep(W_proj, b_proj, nullptr, nullptr, nullptr, nullptr, projWT, projFB, 256, 128);
    prep(W_enc1, b_enc1, g1, bb1, m1, v1, enc1WT, enc1FB, 128, 256);
    prep(W_enc2, b_enc2, g2, bb2, m2, v2, enc2WT, enc2FB, 256, 256);
    prep(W_g0,   b_g0,   nullptr, nullptr, nullptr, nullptr, g0WT, g0FB, 256, 256);
    prep(W_g1,   b_g1,   nullptr, nullptr, nullptr, nullptr, g1WT, g1FB, 256, 256);
    prep(W_ta1,  b_ta1,  gt, bbt, mt, vt, ta1WT, ta1FB, 512, 256);
    prep(W_ta2,  b_ta2,  nullptr, nullptr, nullptr, nullptr, ta2WT, ta2FB, 256, 128);
    prep(W_ep1,  b_ep1,  ge, bbe, me, ve, ep1WT, ep1FB, 512, 256);
    prep(W_ep2,  b_ep2,  nullptr, nullptr, nullptr, nullptr, ep2WT, ep2FB, 256, 128);
    prep(W_id1,  b_id1,  nullptr, nullptr, nullptr, nullptr, id1WT, id1FB, 256, 256);
    prep(W_id2,  b_id2,  nullptr, nullptr, nullptr, nullptr, id2WT, id2FB, 256, 128);

    const int gridM = ceilDiv(NNODES, 64);   // 313
    const int NC256 = NNODES * HID;          // 5.12M

    // --- node encoder ---
    f32_to_f16_k<<<ceilDiv(NC256, 256), 256, 0, stream>>>(nf, B0h, NC256);
    gemm_wmma<256, 128, false><<<dim3(gridM, 2), 128, 0, stream>>>(B0h, projWT, projFB, nullptr, B1h, NNODES);
    gemm_wmma<128, 256, true ><<<dim3(gridM, 4), 128, 0, stream>>>(B1h, enc1WT, enc1FB, nullptr, B0h, NNODES);
    gemm_wmma<256, 256, true ><<<dim3(gridM, 4), 128, 0, stream>>>(B0h, enc2WT, enc2FB, B2f, B1h, NNODES);
    // B1h = x2 (f16), B2f = x2 (f32)

    // --- temporal attention head on pre-GCN embeddings ---
    edge_head<<<NEDGES / 16, 256, 0, stream>>>(B1h, src, dst, ta1WT, ta1FB, ta2WT, ta2FB,
                                               W_ta3, b_ta3, attnB, NEDGES);

    // --- degree / norm (shared by both GCN layers) ---
    fill_f32_k<<<ceilDiv(NNODES, 256), 256, 0, stream>>>(degB, 1.0f, NNODES);
    deg_acc_k<<<ceilDiv(NEDGES, 256), 256, 0, stream>>>(dst, degB, NEDGES);
    rsqrt_k<<<ceilDiv(NNODES, 256), 256, 0, stream>>>(degB, dinvB, NNODES);

    // --- GCN layer 1 (+ residual) ---
    gemm_wmma<256, 256, false><<<dim3(gridM, 4), 128, 0, stream>>>(B1h, g0WT, nullptr, B3f, nullptr, NNODES);
    fill_f32_k<<<ceilDiv(NC256, 256), 256, 0, stream>>>(B4f, 0.0f, NC256);
    gcn_scatter<<<ceilDiv(NEDGES * HID, 256), 256, 0, stream>>>(B3f, src, dst, dinvB, B4f, NEDGES);
    gcn_finalize<<<ceilDiv(NC256, 256), 256, 0, stream>>>(B4f, B3f, B2f, dinvB, b_g0, B0f, B1h, NNODES);
    // B0f = x3 (f32), B1h = x3 (f16)

    // --- GCN layer 2 (+ residual) -> final x into d_out ---
    gemm_wmma<256, 256, false><<<dim3(gridM, 4), 128, 0, stream>>>(B1h, g1WT, nullptr, B3f, nullptr, NNODES);
    fill_f32_k<<<ceilDiv(NC256, 256), 256, 0, stream>>>(B4f, 0.0f, NC256);
    gcn_scatter<<<ceilDiv(NEDGES * HID, 256), 256, 0, stream>>>(B3f, src, dst, dinvB, B4f, NEDGES);
    gcn_finalize<<<ceilDiv(NC256, 256), 256, 0, stream>>>(B4f, B3f, B0f, dinvB, b_g1, outX, B2h, NNODES);
    // outX = final x (f32), B2h = final x (f16)

    // --- edge predictor head on final embeddings + temporal boost ---
    edge_head<<<NEDGES / 16, 256, 0, stream>>>(B2h, src, dst, ep1WT, ep1FB, ep2WT, ep2FB,
                                               W_ep3, b_ep3, ewB, NEDGES);
    boost_k<<<ceilDiv(NEDGES, 256), 256, 0, stream>>>(ewB, attnB, src, dst, outEW, NEDGES);

    // --- id classifier ---
    gemm_wmma<256, 256, true><<<dim3(gridM, 4), 128, 0, stream>>>(B2h, id1WT, id1FB, nullptr, B1h, NNODES);
    gemm_wmma<256, 128, true><<<dim3(gridM, 2), 128, 0, stream>>>(B1h, id2WT, id2FB, outID, nullptr, NNODES);
}
